// Attention_72138270704022
// MI455X (gfx1250) — matrix operand order repaired and checked by reading, not compiled, not run on hardware
//
#include <hip/hip_runtime.h>
#include <stdint.h>

// Problem constants from the reference
#define S_LEN 2048
#define B_DIM 32
#define H_DIM 1024
#define M_ROWS (S_LEN * B_DIM)   // 65536 GEMM rows

typedef __attribute__((ext_vector_type(16))) __bf16 v16bf;
typedef __attribute__((ext_vector_type(8)))  float  v8f;
typedef __attribute__((ext_vector_type(8)))  unsigned int v8u;

// float -> bf16 round-to-nearest-even (scalar fallback)
__device__ __forceinline__ unsigned short f2bf(float f) {
    union { float f; unsigned int u; } c; c.f = f;
    unsigned int u = c.u;
    unsigned int r = u + 0x7FFFu + ((u >> 16) & 1u);
    return (unsigned short)(r >> 16);
}

// pack two floats into one dword of bf16 (lo, hi) — prefer v_cvt_pk_bf16_f32
__device__ __forceinline__ unsigned int pack_bf16(float lo, float hi) {
#if __has_builtin(__builtin_amdgcn_cvt_pk_bf16_f32)
    auto pk = __builtin_amdgcn_cvt_pk_bf16_f32(lo, hi);
    return __builtin_bit_cast(unsigned int, pk);
#else
    return ((unsigned int)f2bf(hi) << 16) | (unsigned int)f2bf(lo);
#endif
}

// prefer hardware v_tanh_f32 (gfx1250 trans op)
__device__ __forceinline__ float fast_tanh(float x) {
#if __has_builtin(__builtin_amdgcn_tanhf)
    return __builtin_amdgcn_tanhf(x);
#else
    return tanhf(x);
#endif
}

// ---------------------------------------------------------------------------
// Kernel 1: convert We = W[:, H:] to bf16, layout we[n*H + h]  (h contiguous)
// so each lane's WMMA B-fragment is one contiguous 32B load.
// ---------------------------------------------------------------------------
__global__ void prep_we_kernel(const float* __restrict__ W,
                               unsigned int* __restrict__ we_u) {
    int idx = blockIdx.x * blockDim.x + threadIdx.x;   // 524288 threads, 2 elems each
    int n  = idx >> 9;
    int h2 = idx & 511;
    const float2 f = *reinterpret_cast<const float2*>(
        W + (size_t)n * 2048 + 1024 + h2 * 2);
    we_u[(size_t)n * 512 + h2] = pack_bf16(f.x, f.y);
}

// ---------------------------------------------------------------------------
// Kernel 2: pre_h[b,k] = sum_h hidden[b,h] * W[k,h]   (tiny: 0.07 GFLOP)
// ---------------------------------------------------------------------------
__global__ void preh_kernel(const float* __restrict__ hidden,
                            const float* __restrict__ W,
                            float* __restrict__ preh) {
    int g = blockIdx.x * blockDim.x + threadIdx.x;     // 32768 threads
    int b = g & 31;
    int k = g >> 5;                                    // whole wave shares k
    const float* hrow = hidden + (size_t)b * H_DIM;
    const float* wrow = W + (size_t)k * 2048;
    float acc = 0.f;
    #pragma unroll 8
    for (int h = 0; h < H_DIM; ++h) acc += hrow[h] * wrow[h];
    preh[(size_t)b * H_DIM + k] = acc;
}

// ---------------------------------------------------------------------------
// Kernel 3: fused GEMM + tanh-energy + v-dot.
//   Each workgroup: 32 rows of enc (read once from HBM), full N=1024 sweep.
//   8 waves; wave w owns N-tiles n0 = (w + 8j)*16, j=0..7.
//   TWO A m-tiles share every B fragment (halves L2 traffic, 2 wmma / B-load).
//   Accumulators held across the entire K loop (2 x 8 x v8f = 128 VGPRs).
// ---------------------------------------------------------------------------
#define LDSW 516   // padded row stride in dwords (1024 bf16 + 8 pad) -> conflict-light

__global__ void __launch_bounds__(256)
gemm_energy_kernel(const float* __restrict__ enc,
                   const unsigned short* __restrict__ we,
                   const float* __restrict__ preh,
                   const float* __restrict__ bias,
                   const float* __restrict__ vvec,
                   float* __restrict__ scores) {
    __shared__ unsigned int a_lds[32 * LDSW];   // 32 rows x 1024 bf16 (padded) ~66KB
    __shared__ float sc_lds[8 * 32];            // per-wave partial scores

    const int tid    = threadIdx.x;
    const int m_base = blockIdx.x * 32;

    // ---- stage A tile: 32 rows x 256 float4, converting fp32 -> packed bf16
    #pragma unroll
    for (int i = 0; i < 32; ++i) {
        int idx = tid + i * 256;                 // 0..8191 float4s
        int row = idx >> 8;
        int col = idx & 255;                     // float4 column
        const float4 f = *reinterpret_cast<const float4*>(
            enc + (size_t)(m_base + row) * H_DIM + col * 4);
        uint2 p;
        p.x = pack_bf16(f.x, f.y);
        p.y = pack_bf16(f.z, f.w);
        *reinterpret_cast<uint2*>(&a_lds[row * LDSW + col * 2]) = p;
    }
    __syncthreads();

    const int wave = tid >> 5;
    const int lane = tid & 31;
    const int half = lane >> 4;     // which 16-lane half
    const int nlan = lane & 15;     // N index within tile / A row (M)

    v8f acc[2][8];
    #pragma unroll
    for (int t = 0; t < 2; ++t)
        #pragma unroll
        for (int j = 0; j < 8; ++j)
            acc[t][j] = (v8f){0.f,0.f,0.f,0.f,0.f,0.f,0.f,0.f};

    // ---- K loop: 32 steps of K=32
    for (int ks = 0; ks < 32; ++ks) {
        const int k0 = ks * 32;
        // A fragments, contiguous k order (same as B):
        //   lane: M = nlan; V0..7 -> K = half*16 + {0..15}
        const unsigned int* ab0 = &a_lds[nlan * LDSW + (k0 >> 1) + half * 8];
        const unsigned int* ab1 = ab0 + 16 * LDSW;
        const uint4 p0 = *reinterpret_cast<const uint4*>(ab0);
        const uint4 p1 = *reinterpret_cast<const uint4*>(ab0 + 4);
        const uint4 q0 = *reinterpret_cast<const uint4*>(ab1);
        const uint4 q1 = *reinterpret_cast<const uint4*>(ab1 + 4);
        const v16bf a0 = __builtin_bit_cast(
            v16bf, (v8u){p0.x, p0.y, p0.z, p0.w, p1.x, p1.y, p1.z, p1.w});
        const v16bf a1 = __builtin_bit_cast(
            v16bf, (v8u){q0.x, q0.y, q0.z, q0.w, q1.x, q1.y, q1.z, q1.w});

        #pragma unroll
        for (int j = 0; j < 8; ++j) {
            const int n0 = (wave + j * 8) * 16;
            // B fragment, contiguous k order (same as A):
            //   lane: N = n0+nlan; K = half*16 .. half*16+15 (contiguous 32B)
            const v8u bu = *reinterpret_cast<const v8u*>(
                we + (size_t)(n0 + nlan) * H_DIM + k0 + half * 16);
            const v16bf bfrag = __builtin_bit_cast(v16bf, bu);
            acc[0][j] = __builtin_amdgcn_wmma_f32_16x16x32_bf16(
                false, a0, false, bfrag, (short)0, acc[0][j], false, false);
            acc[1][j] = __builtin_amdgcn_wmma_f32_16x16x32_bf16(
                false, a1, false, bfrag, (short)0, acc[1][j], false, false);
        }
    }

    // ---- fused epilogue: energy = tanh(acc + pre_h + b), partial v-dot
    // C/D layout: lane holds column n = n0+nlan, rows M = half*8 + r (+ t*16)
    float part[2][8];
    #pragma unroll
    for (int t = 0; t < 2; ++t)
        #pragma unroll
        for (int r = 0; r < 8; ++r) part[t][r] = 0.f;

    #pragma unroll
    for (int j = 0; j < 8; ++j) {
        const int n  = (wave + j * 8) * 16 + nlan;
        const float bn = bias[n];
        const float vn = vvec[n];
        #pragma unroll
        for (int t = 0; t < 2; ++t) {
            #pragma unroll
            for (int r = 0; r < 8; ++r) {
                // global row = m_base + t*16 + half*8 + r; m_base % 32 == 0
                const int b = t * 16 + half * 8 + r;        // row & 31
                const float ph = preh[(size_t)b * H_DIM + n];
                part[t][r] += fast_tanh(acc[t][j][r] + ph + bn) * vn;
            }
        }
    }

    // ---- reduce across the 16 N-lanes (xor masks stay within each half)
    #pragma unroll
    for (int t = 0; t < 2; ++t) {
        #pragma unroll
        for (int r = 0; r < 8; ++r) {
            float x = part[t][r];
            x += __shfl_xor(x, 1, 32);
            x += __shfl_xor(x, 2, 32);
            x += __shfl_xor(x, 4, 32);
            x += __shfl_xor(x, 8, 32);
            if (nlan == 0) sc_lds[wave * 32 + t * 16 + half * 8 + r] = x;
        }
    }
    __syncthreads();

    // ---- deterministic cross-wave reduction, single store per row
    if (tid < 32) {
        float s = 0.f;
        #pragma unroll
        for (int w = 0; w < 8; ++w) s += sc_lds[w * 32 + tid];
        scores[m_base + tid] = s;
    }
}

// ---------------------------------------------------------------------------
// Kernel 4: softmax over s for each b. scores indexed [s*B + b]; out [b*S + s].
// ---------------------------------------------------------------------------
__global__ void __launch_bounds__(256)
softmax_kernel(const float* __restrict__ scores, float* __restrict__ out) {
    const int b   = blockIdx.x;     // 32 batches
    const int tid = threadIdx.x;    // 256 threads, 8 elems each
    __shared__ float red[256];

    float vals[8];
    float mx = -INFINITY;
    #pragma unroll
    for (int i = 0; i < 8; ++i) {
        const int s = tid + i * 256;
        vals[i] = scores[(size_t)s * B_DIM + b];
        mx = fmaxf(mx, vals[i]);
    }
    red[tid] = mx;
    __syncthreads();
    for (int st = 128; st > 0; st >>= 1) {
        if (tid < st) red[tid] = fmaxf(red[tid], red[tid + st]);
        __syncthreads();
    }
    mx = red[0];
    __syncthreads();

    float sum = 0.f;
    #pragma unroll
    for (int i = 0; i < 8; ++i) {
        vals[i] = __expf(vals[i] - mx);
        sum += vals[i];
    }
    red[tid] = sum;
    __syncthreads();
    for (int st = 128; st > 0; st >>= 1) {
        if (tid < st) red[tid] += red[tid + st];
        __syncthreads();
    }
    const float inv = 1.0f / red[0];
    #pragma unroll
    for (int i = 0; i < 8; ++i)
        out[(size_t)b * S_LEN + tid + i * 256] = vals[i] * inv;
}

// ---------------------------------------------------------------------------
extern "C" void kernel_launch(void* const* d_in, const int* in_sizes, int n_in,
                              void* d_out, int out_size, void* d_ws, size_t ws_size,
                              hipStream_t stream) {
    const float* hidden = (const float*)d_in[0];   // (1,B,H)
    const float* enc    = (const float*)d_in[1];   // (S,B,H)
    const float* W      = (const float*)d_in[2];   // (H,2H)
    const float* bias   = (const float*)d_in[3];   // (H,)
    const float* vvec   = (const float*)d_in[4];   // (H,)
    float* out = (float*)d_out;                    // (B,1,S)

    // workspace layout: [we bf16: 2MB][pre_h: 128KB][scores: 256KB]
    char* ws = (char*)d_ws;
    unsigned int*   we_u   = (unsigned int*)ws;
    unsigned short* we     = (unsigned short*)ws;
    float*          preh   = (float*)(ws + (2u << 20));
    float*          scores = (float*)(ws + (2u << 20) + (128u << 10));

    prep_we_kernel<<<(H_DIM * H_DIM / 2) / 256, 256, 0, stream>>>(W, we_u);
    preh_kernel<<<(B_DIM * H_DIM) / 256, 256, 0, stream>>>(hidden, W, preh);
    gemm_energy_kernel<<<M_ROWS / 32, 256, 0, stream>>>(enc, we, preh, bias, vvec, scores);
    softmax_kernel<<<B_DIM, 256, 0, stream>>>(scores, out);
}
